// AttentionLayer_45002667327819
// MI455X (gfx1250) — matrix operand order repaired and checked by reading, not compiled, not run on hardware
//
#include <hip/hip_runtime.h>

// ---------------------------------------------------------------------------
// CDNA5 (gfx1250) fused multi-head attention, bf16 WMMA pipeline.
//   0) convert_bf16 : activations f32 -> bf16 (once; GEMM loops stay pure WMMA)
//   1) transpose_w  : W[k][n] f32 -> Wt[n][k] bf16 (x4 weights)
//   2) gemm_bf16    : A bf16 @ Wt bf16 + bias; epilogue mode selects
//                     bf16 row-major (q,k) / per-head-transposed vt (v) / f32 (out)
//   3) attn         : flash-style softmax(QK^T/8)V, 16 query rows per wave
// All matrix math: v_wmma_f32_16x16x32_bf16 (wave32, f32 accum).
// ---------------------------------------------------------------------------

typedef __attribute__((ext_vector_type(16))) __bf16 v16bf;
typedef __attribute__((ext_vector_type(8)))  float v8f;
typedef __attribute__((ext_vector_type(8)))  unsigned short ushort8;
typedef __attribute__((ext_vector_type(4)))  float float4v;

#define D_MODEL 1024
#define LSEQ    2048
#define NB      4
#define NH      16
#define DH      64

__device__ __forceinline__ unsigned short f2bf(float f) {
  return __builtin_bit_cast(unsigned short, (__bf16)f);  // hardware cvt
}

union Frag {
  ushort8        u8[2];
  unsigned short us[16];
  v16bf          v;
};

__device__ __forceinline__ v8f wmma_bf16(const Frag& a, const Frag& b, v8f c) {
  // D = A(16x32 bf16) * B(32x16 bf16) + C(16x16 f32)
  return __builtin_amdgcn_wmma_f32_16x16x32_bf16(false, a.v, false, b.v,
                                                 (short)0, c, false, false);
}

// ---------------------------------------------------------------------------
// 0) Elementwise f32 -> bf16 (8 elems/thread, 32B in / 16B out)
// ---------------------------------------------------------------------------
__global__ void __launch_bounds__(256) convert_bf16(
    const float* __restrict__ x, unsigned short* __restrict__ y) {
  const int i = (blockIdx.x * 256 + threadIdx.x) * 8;
  float4v a = *(const float4v*)(x + i);
  float4v b = *(const float4v*)(x + i + 4);
  ushort8 o;
#pragma unroll
  for (int j = 0; j < 4; ++j) { o[j] = f2bf(a[j]); o[4 + j] = f2bf(b[j]); }
  *(ushort8*)(y + i) = o;
}

// ---------------------------------------------------------------------------
// 1) Weight transpose + bf16 convert: Wt[n*1024+k] = bf16(W[k*1024+n])
// ---------------------------------------------------------------------------
__global__ void __launch_bounds__(256) transpose_w(
    const float* __restrict__ W0, const float* __restrict__ W1,
    const float* __restrict__ W2, const float* __restrict__ W3,
    unsigned short* __restrict__ T0, unsigned short* __restrict__ T1,
    unsigned short* __restrict__ T2, unsigned short* __restrict__ T3) {
  const float* W; unsigned short* T;
  switch (blockIdx.z) {
    case 0:  W = W0; T = T0; break;
    case 1:  W = W1; T = T1; break;
    case 2:  W = W2; T = T2; break;
    default: W = W3; T = T3; break;
  }
  int idx = blockIdx.x * 256 + threadIdx.x;      // 1M elements
  int k = idx >> 10, n = idx & 1023;
  T[n * D_MODEL + k] = f2bf(W[idx]);
}

// ---------------------------------------------------------------------------
// 2) Unified GEMM: C[8192,1024] = A bf16 @ Wt bf16 + bias.
//    Block = 256 thr = 8 waves (4M x 2N); block tile 128x128; wave 32x64
//    => per k-step: 2 A-frags + 4 B-frags (12x 16B loads) feed 8 WMMAs.
//    mode 0: bf16 out row-major        (q, k projections)
//    mode 1: bf16 out vt[b][h][e][s]   (v projection, pre-transposed for P@V)
//    mode 2: f32 out row-major         (final output projection)
// ---------------------------------------------------------------------------
__global__ void __launch_bounds__(256) gemm_bf16_kernel(
    const unsigned short* __restrict__ A,   // [8192,1024] bf16
    const unsigned short* __restrict__ Wt,  // [1024(n),1024(k)] bf16
    const float* __restrict__ bias,         // [1024]
    void* __restrict__ outv, int mode) {
  const int tid = threadIdx.x, lane = tid & 31, wid = tid >> 5;
  const int ln = lane & 15, hi = lane >> 4;
  const int wm = wid & 3, wn = wid >> 2;
  const int m0 = blockIdx.x * 128 + wm * 32;   // 2 x 16-row subtiles
  const int n0 = blockIdx.y * 128 + wn * 64;   // 4 x 16-col subtiles

  v8f acc[2][4];
#pragma unroll
  for (int mi = 0; mi < 2; ++mi)
#pragma unroll
    for (int ni = 0; ni < 4; ++ni)
#pragma unroll
      for (int r = 0; r < 8; ++r) acc[mi][ni][r] = 0.0f;

  for (int k0 = 0; k0 < D_MODEL; k0 += 32) {
    Frag af[2], bfr[4];
#pragma unroll
    for (int mi = 0; mi < 2; ++mi) {
      // A-frag (16x32): row = ln; elems 0..7 at K=k0+8*hi, elems 8..15 at +16
      const unsigned short* ap =
          A + (size_t)(m0 + mi * 16 + ln) * D_MODEL + k0 + 8 * hi;
      af[mi].u8[0] = *(const ushort8*)(ap);
      af[mi].u8[1] = *(const ushort8*)(ap + 16);
    }
#pragma unroll
    for (int ni = 0; ni < 4; ++ni) {
      // B-frag (32x16): col = n; elems 0..7 at K=k0+8*hi, elems 8..15 at +16
      const unsigned short* bp =
          Wt + (size_t)(n0 + ni * 16 + ln) * D_MODEL + k0 + 8 * hi;
      bfr[ni].u8[0] = *(const ushort8*)(bp);
      bfr[ni].u8[1] = *(const ushort8*)(bp + 16);
    }
#pragma unroll
    for (int mi = 0; mi < 2; ++mi)
#pragma unroll
      for (int ni = 0; ni < 4; ++ni)
        acc[mi][ni] = wmma_bf16(af[mi], bfr[ni], acc[mi][ni]);
  }

#pragma unroll
  for (int mi = 0; mi < 2; ++mi)
#pragma unroll
    for (int ni = 0; ni < 4; ++ni) {
      const int col = n0 + ni * 16 + ln;
      const float bv_ = bias[col];
#pragma unroll
      for (int r = 0; r < 8; ++r) {
        const int row = m0 + mi * 16 + r + 8 * hi;  // C layout: M = r + 8*hi
        const float v = acc[mi][ni][r] + bv_;
        if (mode == 2) {
          ((float*)outv)[(size_t)row * D_MODEL + col] = v;
        } else if (mode == 0) {
          ((unsigned short*)outv)[(size_t)row * D_MODEL + col] = f2bf(v);
        } else {
          int b = row >> 11, s = row & 2047, h = col >> 6, e = col & 63;
          ((unsigned short*)outv)[(size_t)((b * NH + h) * DH + e) * LSEQ + s] =
              f2bf(v);
        }
      }
    }
}

// ---------------------------------------------------------------------------
// 3) Attention: each wave owns 16 query rows of one (b,h); streams S in
//    chunks of 32 with online softmax; P-tile round-trips through LDS to
//    convert C-layout -> A-layout for the P@V WMMA.
// ---------------------------------------------------------------------------
__global__ void __launch_bounds__(256) attn_kernel(
    const unsigned short* __restrict__ q,   // [8192][1024] bf16
    const unsigned short* __restrict__ k,   // [8192][1024] bf16
    const unsigned short* __restrict__ vt,  // [(b*16+h)*64+e][2048] bf16
    unsigned short* __restrict__ o) {       // [(b*16+h)*2048+l][64] bf16
  __shared__ unsigned short plds[8 * 16 * 32];
  const int tid = threadIdx.x, lane = tid & 31, wid = tid >> 5;
  const int ln = lane & 15, hi = lane >> 4;
  const int b = blockIdx.z, h = blockIdx.y;
  const int l0 = blockIdx.x * 128 + wid * 16;
  unsigned short* pw = plds + wid * (16 * 32);

  // Q fragments for e = 0..31 and 32..63
  Frag qa[2];
  {
    const unsigned short* qp =
        q + (size_t)(b * LSEQ + l0 + ln) * D_MODEL + h * DH + 8 * hi;
    qa[0].u8[0] = *(const ushort8*)(qp);
    qa[0].u8[1] = *(const ushort8*)(qp + 16);
    qa[1].u8[0] = *(const ushort8*)(qp + 32);
    qa[1].u8[1] = *(const ushort8*)(qp + 48);
  }

  v8f oacc[4];
  float m[8], l[8];
#pragma unroll
  for (int ni = 0; ni < 4; ++ni)
#pragma unroll
    for (int r = 0; r < 8; ++r) oacc[ni][r] = 0.0f;
#pragma unroll
  for (int r = 0; r < 8; ++r) { m[r] = -3.0e38f; l[r] = 0.0f; }

  const float scale = 0.125f;  // 1/sqrt(64)

  for (int s0 = 0; s0 < LSEQ; s0 += 32) {
    // --- scores: two 16x16 tiles over the 32 S-columns ---
    v8f st[2];
#pragma unroll
    for (int t = 0; t < 2; ++t) {
      Frag kb0, kb1;  // B-frag: col = s, K = e (contiguous in k row)
      const unsigned short* kp =
          k + (size_t)(b * LSEQ + s0 + t * 16 + ln) * D_MODEL + h * DH + 8 * hi;
      kb0.u8[0] = *(const ushort8*)(kp);
      kb0.u8[1] = *(const ushort8*)(kp + 16);
      kb1.u8[0] = *(const ushort8*)(kp + 32);
      kb1.u8[1] = *(const ushort8*)(kp + 48);
      v8f sc;
#pragma unroll
      for (int r = 0; r < 8; ++r) sc[r] = 0.0f;
      sc = wmma_bf16(qa[0], kb0, sc);
      sc = wmma_bf16(qa[1], kb1, sc);
      st[t] = sc;
    }

    // --- online softmax (rows split per 16-lane half: M = r + 8*hi) ---
    float mt[8];
#pragma unroll
    for (int r = 0; r < 8; ++r) mt[r] = fmaxf(st[0][r], st[1][r]);
#pragma unroll
    for (int off = 1; off < 16; off <<= 1)
#pragma unroll
      for (int r = 0; r < 8; ++r) mt[r] = fmaxf(mt[r], __shfl_xor(mt[r], off, 32));

    float corr[8], p0[8], p1[8], rs[8];
#pragma unroll
    for (int r = 0; r < 8; ++r) {
      float mn = fmaxf(m[r], mt[r] * scale);
      corr[r] = __expf(m[r] - mn);
      m[r] = mn;
      p0[r] = __expf(st[0][r] * scale - mn);
      p1[r] = __expf(st[1][r] * scale - mn);
      rs[r] = p0[r] + p1[r];
    }
#pragma unroll
    for (int off = 1; off < 16; off <<= 1)
#pragma unroll
      for (int r = 0; r < 8; ++r) rs[r] += __shfl_xor(rs[r], off, 32);
#pragma unroll
    for (int r = 0; r < 8; ++r) l[r] = l[r] * corr[r] + rs[r];
#pragma unroll
    for (int ni = 0; ni < 4; ++ni)
#pragma unroll
      for (int r = 0; r < 8; ++r) oacc[ni][r] = oacc[ni][r] * corr[r];

    // --- P tile: C-layout -> LDS -> A-layout bf16 fragment ---
#pragma unroll
    for (int r = 0; r < 8; ++r) {
      pw[(r + 8 * hi) * 32 + 0  + ln] = f2bf(p0[r]);
      pw[(r + 8 * hi) * 32 + 16 + ln] = f2bf(p1[r]);
    }
    asm volatile("s_wait_dscnt 0x0" ::: "memory");
    Frag pa;
    pa.u8[0] = *(const ushort8*)(pw + ln * 32 + 8 * hi);
    pa.u8[1] = *(const ushort8*)(pw + ln * 32 + 8 * hi + 16);

    // --- O += P(16x32) @ V(32x64), vt gives contiguous B-fragments ---
#pragma unroll
    for (int ni = 0; ni < 4; ++ni) {
      Frag vb;
      const unsigned short* vp =
          vt + (size_t)((b * NH + h) * DH + ni * 16 + ln) * LSEQ + s0 + 8 * hi;
      vb.u8[0] = *(const ushort8*)(vp);
      vb.u8[1] = *(const ushort8*)(vp + 16);
      oacc[ni] = wmma_bf16(pa, vb, oacc[ni]);
    }
  }

  // --- normalize and store: o[b][h][l][e] bf16 (== reshaped GEMM input) ---
  float inv[8];
#pragma unroll
  for (int r = 0; r < 8; ++r) inv[r] = 1.0f / l[r];
#pragma unroll
  for (int ni = 0; ni < 4; ++ni)
#pragma unroll
    for (int r = 0; r < 8; ++r) {
      size_t idx = (size_t)((b * NH + h) * LSEQ + l0 + r + 8 * hi) * DH + ni * 16 + ln;
      o[idx] = f2bf(oacc[ni][r] * inv[r]);
    }
}

// ---------------------------------------------------------------------------
extern "C" void kernel_launch(void* const* d_in, const int* in_sizes, int n_in,
                              void* d_out, int out_size, void* d_ws, size_t ws_size,
                              hipStream_t stream) {
  const float* queries = (const float*)d_in[0];
  const float* keys    = (const float*)d_in[1];
  const float* values  = (const float*)d_in[2];
  const float* Wq = (const float*)d_in[3];
  const float* bq = (const float*)d_in[4];
  const float* Wk = (const float*)d_in[5];
  const float* bk = (const float*)d_in[6];
  const float* Wv = (const float*)d_in[7];
  const float* bv = (const float*)d_in[8];
  const float* Wo = (const float*)d_in[9];
  const float* bo = (const float*)d_in[10];

  const size_t WSZ = (size_t)D_MODEL * D_MODEL;   // 1M elems
  const size_t ASZ = (size_t)NB * LSEQ * D_MODEL; // 8M elems
  unsigned short* wtq   = (unsigned short*)d_ws;
  unsigned short* wtk   = wtq + WSZ;
  unsigned short* wtv   = wtk + WSZ;
  unsigned short* wto   = wtv + WSZ;
  unsigned short* q_ws  = wto + WSZ;
  unsigned short* k_ws  = q_ws + ASZ;
  unsigned short* vt_ws = k_ws + ASZ;
  unsigned short* o_ws  = vt_ws + ASZ;
  unsigned short* xbuf  = o_ws + ASZ;             // staging, reused; total 88 MB

  const dim3 blk(256);
  const dim3 gcv(ASZ / (256 * 8));                // 4096
  const dim3 ggm(64, 8);                          // 128x128 tiles over 8192x1024

  transpose_w<<<dim3(4096, 1, 4), blk, 0, stream>>>(Wq, Wk, Wv, Wo,
                                                    wtq, wtk, wtv, wto);

  convert_bf16<<<gcv, blk, 0, stream>>>(queries, xbuf);
  gemm_bf16_kernel<<<ggm, blk, 0, stream>>>(xbuf, wtq, bq, q_ws, 0);

  convert_bf16<<<gcv, blk, 0, stream>>>(keys, xbuf);
  gemm_bf16_kernel<<<ggm, blk, 0, stream>>>(xbuf, wtk, bk, k_ws, 0);

  convert_bf16<<<gcv, blk, 0, stream>>>(values, xbuf);
  gemm_bf16_kernel<<<ggm, blk, 0, stream>>>(xbuf, wtv, bv, vt_ws, 1);

  attn_kernel<<<dim3(16, 16, 4), blk, 0, stream>>>(q_ws, k_ws, vt_ws, o_ws);

  gemm_bf16_kernel<<<ggm, blk, 0, stream>>>(o_ws, wto, bo, d_out, 2);
}
